// AWD_LSTM_82343112999184
// MI455X (gfx1250) — matrix operand order repaired and checked by reading, not compiled, not run on hardware
//
#include <hip/hip_runtime.h>
#include <math.h>

// ---------------------------------------------------------------------------
// AWD-LSTM forward for MI455X (gfx1250, wave32, WMMA).
//
// ~1.8 TFLOP of GEMM work; weights (~27MB bf16) are L2-resident (192MB L2),
// so the recurrence is matrix-pipe bound -> v_wmma_f32_16x16x32_bf16.
// Only layer0 is sequential (layer1 consumes flipped layer0 state, not its
// own), so layer1 runs as large batched GEMMs (M=2048) after the scan.
// A-tiles are staged to LDS with GLOBAL_LOAD_ASYNC_TO_LDS_B128 (ASYNCcnt),
// double-buffered with a peeled, branch-free steady-state loop; B fragments
// stream from L2 with global_prefetch_b8 running one chunk ahead.
// ---------------------------------------------------------------------------

#define T_STEPS 256
#define BSZ     256
#define HID     1024
#define FH      4096      // 4*H
#define FEATP   320       // 319 padded to multiple of 64
#define NCLS    13

typedef __attribute__((ext_vector_type(16))) __bf16        v16bf;
typedef __attribute__((ext_vector_type(8)))  float         v8f;
typedef __attribute__((ext_vector_type(4)))  unsigned int  uint4r;  // raw, union-safe

union Frag16 { v16bf v; uint4r q[2]; };

__device__ __forceinline__ unsigned short f2bf(float x) {
    unsigned int u = __float_as_uint(x);
    u += 0x7FFFu + ((u >> 16) & 1u);          // round-to-nearest-even
    return (unsigned short)(u >> 16);
}
__device__ __forceinline__ float bf2f(unsigned short b) {
    return __uint_as_float(((unsigned int)b) << 16);
}
__device__ __forceinline__ float sigf(float x) {
    return 1.0f / (1.0f + __expf(-x));
}

// ---------------------------------------------------------------------------
// bf16 GEMM:  Out[M x 4096] (+)= A[M x K] * W[4096 x K]^T  (fp32 accumulate)
// Block: 256 threads = 8 waves (2 M-waves x 4 N-waves). Block tile 64x128.
// Wave tile 32x32 = 4 WMMA accumulators; K-chunk = 64 -> 8 WMMA per barrier.
// A tile (64x64 bf16, 8KB) staged via async global->LDS, double-buffered:
// each thread copies 32B (2 x b128), i.e. 2 async ops per wave per chunk.
// flip!=0: A row r -> (r/flip)*flip + flip-1 - r%flip   ([::-1] batch flip).
// accum==0: Out = gemm + b1[n] + b2[n];  accum==1: Out += gemm.
// K must be a multiple of 64 (320 and 1024 both are).
// ---------------------------------------------------------------------------
__global__ __launch_bounds__(256)
void gemm_bf16_nt(const unsigned short* __restrict__ A, int lda, int flip,
                  const unsigned short* __restrict__ W, int K,
                  float* __restrict__ Out, int ldo,
                  const float* __restrict__ b1, const float* __restrict__ b2,
                  int accum)
{
    __shared__ unsigned short sA[2][64][64];

    const int tid   = threadIdx.x;
    const int wave  = tid >> 5;
    const int lane  = tid & 31;
    const int lh    = lane >> 4;      // lane half
    const int l16   = lane & 15;
    const int waveM = wave >> 2;      // 0..1  -> 2 x 32 = 64 rows
    const int waveN = wave & 3;       // 0..3  -> 4 x 32 = 128 cols
    const int m0 = blockIdx.y * 64;
    const int n0 = blockIdx.x * 128 + waveN * 32;

    // Async staging assignment: each thread owns one 32B slice of the tile.
    const int srow = tid >> 2;        // 0..63
    const int scol = (tid & 3) * 16;  // 0,16,32,48
    int grow = m0 + srow;
    if (flip) grow = (grow / flip) * flip + (flip - 1) - (grow % flip);
    const unsigned short* Ag = A + (size_t)grow * lda + scol;

    // LDS aperture keeps the allocation offset in addr[31:0] (ISA 10.2).
    const unsigned ldsoff0 = (unsigned)(unsigned long long)(const void*)&sA[0][srow][scol];
    const unsigned ldsoff1 = (unsigned)(unsigned long long)(const void*)&sA[1][srow][scol];

    auto stage = [&](unsigned ldsoff, const unsigned short* g) {
        asm volatile("global_load_async_to_lds_b128 %0, %1, off"
                     :: "v"(ldsoff), "v"((unsigned long long)g) : "memory");
        asm volatile("global_load_async_to_lds_b128 %0, %1, off"
                     :: "v"(ldsoff + 16u), "v"((unsigned long long)(g + 8)) : "memory");
    };

    v8f acc00 = {}, acc01 = {}, acc10 = {}, acc11 = {};

    auto compute_chunk = [&](int cur, int k0, bool pf) {
        #pragma unroll
        for (int ks = 0; ks < 2; ++ks) {
            const int kk = k0 + ks * 32;

            // A fragments (16x32 each, ISA 7.12.2): row = lane%16;
            // lanes0-15: K 0..7 & 16..23 ; lanes16-31: K 8..15 & 24..31
            Frag16 fa0, fa1;
            const unsigned short* ar0 = &sA[cur][waveM * 32 + l16][ks * 32];
            const unsigned short* ar1 = ar0 + 16 * 64;
            fa0.q[0] = *(const uint4r*)(ar0 + lh * 8);
            fa0.q[1] = *(const uint4r*)(ar0 + 16 + lh * 8);
            fa1.q[0] = *(const uint4r*)(ar1 + lh * 8);
            fa1.q[1] = *(const uint4r*)(ar1 + 16 + lh * 8);

            // B fragments (32x16): col N = lane%16 (= weight row);
            // lanes0-15: K 0..7 & 16..23 ; lanes16-31: K 8..15 & 24..31
            Frag16 fb0, fb1;
            const unsigned short* wp0 = W + (size_t)(n0 + l16)      * K + kk + lh * 8;
            const unsigned short* wp1 = W + (size_t)(n0 + 16 + l16) * K + kk + lh * 8;
            fb0.q[0] = *(const uint4r*)(wp0);
            fb0.q[1] = *(const uint4r*)(wp0 + 16);
            fb1.q[0] = *(const uint4r*)(wp1);
            fb1.q[1] = *(const uint4r*)(wp1 + 16);
            if (pf) {                            // -> global_prefetch_b8
                __builtin_prefetch(wp0 + 64, 0, 1);
                __builtin_prefetch(wp1 + 64, 0, 1);
            }

            acc00 = __builtin_amdgcn_wmma_f32_16x16x32_bf16(
                        false, fa0.v, false, fb0.v, (short)0, acc00, false, false);
            acc01 = __builtin_amdgcn_wmma_f32_16x16x32_bf16(
                        false, fa0.v, false, fb1.v, (short)0, acc01, false, false);
            acc10 = __builtin_amdgcn_wmma_f32_16x16x32_bf16(
                        false, fa1.v, false, fb0.v, (short)0, acc10, false, false);
            acc11 = __builtin_amdgcn_wmma_f32_16x16x32_bf16(
                        false, fa1.v, false, fb1.v, (short)0, acc11, false, false);
        }
    };

    const int nk = K >> 6;            // 64-wide K chunks

    // Prologue: stage chunk 0 into buffer 0 (ASYNCcnt += 2).
    stage(ldsoff0, Ag);

    // Steady state: branch-free. Stage chunk kc+1, wait for chunk kc (our two
    // oldest async ops; per-wave async completion is in-order), compute.
    for (int kc = 0; kc < nk - 1; ++kc) {
        const int cur = kc & 1;
        stage(cur ? ldsoff0 : ldsoff1, Ag + (size_t)(kc + 1) * 64);
        asm volatile("s_wait_asynccnt 0x2" ::: "memory");
        __syncthreads();                       // buf[cur] fully in LDS
        compute_chunk(cur, kc * 64, true);
        __syncthreads();                       // buf[cur] free for re-stage
    }

    // Epilogue: last chunk, nothing left in flight afterwards.
    asm volatile("s_wait_asynccnt 0x0" ::: "memory");
    __syncthreads();
    compute_chunk((nk - 1) & 1, (nk - 1) * 64, false);

    // C/D layout: VGPR r -> M = r (lanes0-15) / r+8 (lanes16-31); N = lane%16
    const int oc0 = n0 + l16;
    const int oc1 = n0 + 16 + l16;
    #pragma unroll
    for (int i = 0; i < 2; ++i) {
        const v8f aj0 = i ? acc10 : acc00;
        const v8f aj1 = i ? acc11 : acc01;
        const int orow = m0 + waveM * 32 + i * 16 + lh * 8;
        #pragma unroll
        for (int r = 0; r < 8; ++r) {
            float v0 = aj0[r];
            float v1 = aj1[r];
            float* p0 = Out + (size_t)(orow + r) * ldo + oc0;
            float* p1 = Out + (size_t)(orow + r) * ldo + oc1;
            if (accum) {
                v0 += *p0;
                v1 += *p1;
            } else {
                v0 += (b1 ? b1[oc0] : 0.f) + (b2 ? b2[oc0] : 0.f);
                v1 += (b1 ? b1[oc1] : 0.f) + (b2 ? b2[oc1] : 0.f);
            }
            *p0 = v0;
            *p1 = v1;
        }
    }
}

// ---------------------------------------------------------------------------
// LSTM gates: pre[rows x 4096] -> h (bf16), c (fp32, optional).
// cflip!=0: c_prev row is batch-flipped within each group of cflip rows.
// ---------------------------------------------------------------------------
__global__ __launch_bounds__(256)
void lstm_gates(const float* __restrict__ pre, const float* __restrict__ cprev,
                int cflip, unsigned short* __restrict__ hout,
                float* __restrict__ cout, int rows)
{
    int idx = blockIdx.x * blockDim.x + threadIdx.x;
    if (idx >= rows * HID) return;
    int r = idx >> 10;
    int k = idx & (HID - 1);
    int crow = r;
    if (cflip) crow = (r / cflip) * cflip + (cflip - 1) - (r % cflip);

    const float* pr = pre + (size_t)r * FH;
    float i = sigf(pr[k]);
    float f = sigf(pr[HID + k]);
    float o = sigf(pr[2 * HID + k]);
    float g = tanhf(pr[3 * HID + k]);
    float c = f * cprev[(size_t)crow * HID + k] + i * g;
    float h = o * tanhf(c);
    if (cout) cout[(size_t)r * HID + k] = c;
    hout[(size_t)r * HID + k] = f2bf(h);
}

// ---------------------------------------------------------------------------
// Embedding gather + concat(casing,pos) + zero pad -> bf16 [T*B x 320]
// ---------------------------------------------------------------------------
__global__ __launch_bounds__(256)
void embed_concat(const int* __restrict__ tokens, const float* __restrict__ casing,
                  const float* __restrict__ pos, const float* __restrict__ emb,
                  unsigned short* __restrict__ xb, int rows)
{
    int idx = blockIdx.x * blockDim.x + threadIdx.x;
    if (idx >= rows * FEATP) return;
    int r = idx / FEATP;
    int c = idx % FEATP;
    float v;
    if (c < 300)      v = emb[(size_t)tokens[r] * 300 + c];
    else if (c < 307) v = casing[(size_t)r * 7 + (c - 300)];
    else if (c < 319) v = pos[(size_t)r * 12 + (c - 307)];
    else              v = 0.f;
    xb[idx] = f2bf(v);
}

__global__ __launch_bounds__(256)
void f32_to_bf16(const float* __restrict__ src, unsigned short* __restrict__ dst, int n)
{
    int i = blockIdx.x * blockDim.x + threadIdx.x;
    if (i < n) dst[i] = f2bf(src[i]);
}

// wi0 [4096 x 319] fp32 -> [4096 x 320] bf16, zero-padded
__global__ __launch_bounds__(256)
void pad_wi0(const float* __restrict__ src, unsigned short* __restrict__ dst)
{
    int i = blockIdx.x * blockDim.x + threadIdx.x;
    if (i >= FH * FEATP) return;
    int r = i / FEATP;
    int c = i % FEATP;
    dst[i] = (c < 319) ? f2bf(src[(size_t)r * 319 + c]) : (unsigned short)0;
}

// ---------------------------------------------------------------------------
// Decoder: out[r, 0..12] = concat(h0[r], h1[r]) . dec_w[c]^T + dec_b[c]
// One wave per row; lanes stride K, shuffle-reduce 13 accumulators.
// ---------------------------------------------------------------------------
__global__ __launch_bounds__(256)
void decoder_kernel(const unsigned short* __restrict__ h0b,
                    const unsigned short* __restrict__ h1b,
                    const float* __restrict__ dw, const float* __restrict__ db,
                    float* __restrict__ out, int rows)
{
    int gwave = (blockIdx.x * blockDim.x + threadIdx.x) >> 5;
    int lane  = threadIdx.x & 31;
    if (gwave >= rows) return;

    float acc[NCLS];
    #pragma unroll
    for (int c = 0; c < NCLS; ++c) acc[c] = 0.f;

    const unsigned short* x0 = h0b + (size_t)gwave * HID;
    const unsigned short* x1 = h1b + (size_t)gwave * HID;
    for (int k = lane; k < 2 * HID; k += 32) {
        float x = (k < HID) ? bf2f(x0[k]) : bf2f(x1[k - HID]);
        #pragma unroll
        for (int c = 0; c < NCLS; ++c) acc[c] += x * dw[c * 2 * HID + k];
    }
    #pragma unroll
    for (int c = 0; c < NCLS; ++c)
        for (int off = 16; off > 0; off >>= 1)
            acc[c] += __shfl_down(acc[c], off, 32);
    if (lane == 0) {
        #pragma unroll
        for (int c = 0; c < NCLS; ++c)
            out[(size_t)gwave * NCLS + c] = acc[c] + db[c];
    }
}

// ---------------------------------------------------------------------------
extern "C" void kernel_launch(void* const* d_in, const int* in_sizes, int n_in,
                              void* d_out, int out_size, void* d_ws, size_t ws_size,
                              hipStream_t stream)
{
    const int*   tokens = (const int*)  d_in[0];
    const float* casing = (const float*)d_in[1];
    const float* pos    = (const float*)d_in[2];
    const float* emb    = (const float*)d_in[3];
    const float* wi0    = (const float*)d_in[4];
    const float* bi0    = (const float*)d_in[5];
    const float* wh0    = (const float*)d_in[6];
    const float* bh0    = (const float*)d_in[7];
    const float* wi1    = (const float*)d_in[8];
    const float* bi1    = (const float*)d_in[9];
    const float* wh1    = (const float*)d_in[10];
    const float* bh1    = (const float*)d_in[11];
    const float* dec_w  = (const float*)d_in[12];
    const float* dec_b  = (const float*)d_in[13];
    const float* h_init = (const float*)d_in[14];   // [2,256,1024]; plane 0 used
    const float* c_init = (const float*)d_in[15];   // [2,256,1024]; plane 0 used
    float* out = (float*)d_out;

    // ---- workspace carve-up (offsets 256B-aligned) ----
    char* ws = (char*)d_ws;
    size_t off = 0;
    auto take = [&](size_t bytes) { char* p = ws + off; off += (bytes + 255) & ~(size_t)255; return p; };

    unsigned short* wi0b  = (unsigned short*)take((size_t)FH * FEATP * 2);
    unsigned short* wh0b  = (unsigned short*)take((size_t)FH * HID * 2);
    unsigned short* wi1b  = (unsigned short*)take((size_t)FH * HID * 2);
    unsigned short* wh1b  = (unsigned short*)take((size_t)FH * HID * 2);
    unsigned short* hinitb= (unsigned short*)take((size_t)BSZ * HID * 2);
    unsigned short* xb    = (unsigned short*)take((size_t)T_STEPS * BSZ * FEATP * 2);
    unsigned short* h0b   = (unsigned short*)take((size_t)T_STEPS * BSZ * HID * 2);
    unsigned short* h1b   = (unsigned short*)take((size_t)T_STEPS * BSZ * HID * 2);
    float*          c0a   = (float*)         take((size_t)T_STEPS * BSZ * HID * 4);
    float*          pre0  = (float*)         take((size_t)BSZ * FH * 4);
    float*          pre1  = (float*)         take((size_t)8 * BSZ * FH * 4);

    dim3 blk(256);

    // ---- weight / state conversion to bf16 ----
    pad_wi0    <<<(FH * FEATP + 255) / 256, blk, 0, stream>>>(wi0, wi0b);
    f32_to_bf16<<<(FH * HID  + 255) / 256, blk, 0, stream>>>(wh0, wh0b, FH * HID);
    f32_to_bf16<<<(FH * HID  + 255) / 256, blk, 0, stream>>>(wi1, wi1b, FH * HID);
    f32_to_bf16<<<(FH * HID  + 255) / 256, blk, 0, stream>>>(wh1, wh1b, FH * HID);
    f32_to_bf16<<<(BSZ * HID + 255) / 256, blk, 0, stream>>>(h_init, hinitb, BSZ * HID);

    // ---- embedding + feature concat ----
    embed_concat<<<(T_STEPS * BSZ * FEATP + 255) / 256, blk, 0, stream>>>(
        tokens, casing, pos, emb, xb, T_STEPS * BSZ);

    // ---- layer 0: sequential scan over T ----
    for (int t = 0; t < T_STEPS; ++t) {
        const unsigned short* xt = xb + (size_t)t * BSZ * FEATP;
        const unsigned short* hp = (t == 0) ? hinitb : h0b + (size_t)(t - 1) * BSZ * HID;
        const float*          cp = (t == 0) ? c_init : c0a + (size_t)(t - 1) * BSZ * HID;
        dim3 g1(FH / 128, BSZ / 64);
        gemm_bf16_nt<<<g1, blk, 0, stream>>>(xt, FEATP, 0, wi0b, FEATP, pre0, FH, bi0, bh0, 0);
        gemm_bf16_nt<<<g1, blk, 0, stream>>>(hp, HID,   0, wh0b, HID,   pre0, FH, nullptr, nullptr, 1);
        lstm_gates<<<(BSZ * HID + 255) / 256, blk, 0, stream>>>(
            pre0, cp, 0, h0b + (size_t)t * BSZ * HID, c0a + (size_t)t * BSZ * HID, BSZ);
    }

    // ---- layer 1: fully batched (no self-recurrence); chunks of 8 steps ----
    for (int t0 = 0; t0 < T_STEPS; t0 += 8) {
        const unsigned short* hc = h0b + (size_t)t0 * BSZ * HID;
        dim3 g2(FH / 128, (8 * BSZ) / 64);
        gemm_bf16_nt<<<g2, blk, 0, stream>>>(hc, HID, 0,   wi1b, HID, pre1, FH, bi1, bh1, 0);
        gemm_bf16_nt<<<g2, blk, 0, stream>>>(hc, HID, BSZ, wh1b, HID, pre1, FH, nullptr, nullptr, 1);
        lstm_gates<<<(8 * BSZ * HID + 255) / 256, blk, 0, stream>>>(
            pre1, c0a + (size_t)t0 * BSZ * HID, BSZ,
            h1b + (size_t)t0 * BSZ * HID, nullptr, 8 * BSZ);
    }

    // ---- decoder ----
    decoder_kernel<<<(T_STEPS * BSZ * 32 + 255) / 256, blk, 0, stream>>>(
        h0b, h1b, dec_w, dec_b, out, T_STEPS * BSZ);
}
